// NeuralMemory_61864708932067
// MI455X (gfx1250) — matrix operand order repaired and checked by reading, not compiled, not run on hardware
//
#include <hip/hip_runtime.h>

typedef unsigned short u16;
typedef __attribute__((ext_vector_type(16))) __bf16 v16bf;
typedef __attribute__((ext_vector_type(8)))  float  v8f;

// Problem constants (match reference)
constexpr int Bsz = 4, S = 1024, D = 512, Mm = 64, T = 1088, BT = 4352;
constexpr int H = 8, HD = 64, WIN = 128;
constexpr float LRc = 1e-3f, WDc = 1e-2f, EPSc = 1e-8f, MAX_ALR = 0.1f;

__device__ __forceinline__ u16 f2bf(float f) {
  union { float f; unsigned u; } v; v.f = f;
  unsigned u = v.u;
  unsigned r = (u + 0x7fffu + ((u >> 16) & 1u)) >> 16;  // RNE
  return (u16)r;
}
__device__ __forceinline__ float bf2f(u16 h) {
  union { unsigned u; float f; } v; v.u = ((unsigned)h) << 16; return v.f;
}
__device__ __forceinline__ float sigm(float x) { return 1.0f / (1.0f + __expf(-x)); }

// ---------------- elementwise kernels ----------------

__global__ void k_cvt(const float* __restrict__ in, u16* __restrict__ out, int n) {
  int i = blockIdx.x * 256 + threadIdx.x;
  if (i < n) out[i] = f2bf(in[i]);
}

// out[a*D+b] = bf16(in[b*D+a])  (512x512 f32 -> transposed bf16)
__global__ void k_cvtT(const float* __restrict__ in, u16* __restrict__ out) {
  int i = blockIdx.x * 256 + threadIdx.x;
  if (i >= D * D) return;
  int a = i / D, b = i % D;
  out[i] = f2bf(in[b * D + a]);
}

// tiled bf16 transpose: in [R][C] -> out [C][R]
__global__ __launch_bounds__(256) void k_tr(const u16* __restrict__ in, u16* __restrict__ out,
                                            int R, int C) {
  __shared__ u16 tile[32][33];
  int c0 = blockIdx.x * 32, r0 = blockIdx.y * 32;
  int tx = threadIdx.x & 31, ty = threadIdx.x >> 5;  // 32 x 8
#pragma unroll
  for (int j = 0; j < 32; j += 8)
    tile[ty + j][tx] = in[(size_t)(r0 + ty + j) * C + c0 + tx];
  __syncthreads();
#pragma unroll
  for (int j = 0; j < 32; j += 8)
    out[(size_t)(c0 + ty + j) * R + r0 + tx] = tile[tx][ty + j];
}

__global__ void k_build_xm(const float* __restrict__ x, const float* __restrict__ meta,
                           u16* __restrict__ xm) {
  int i = blockIdx.x * 256 + threadIdx.x;
  if (i >= BT * D) return;
  int r = i / D, c = i % D;
  int b = r / T, t = r % T;
  float v = (t < Mm) ? meta[(size_t)t * D + c]
                     : x[((size_t)(b * S + (t - Mm))) * D + c];
  xm[i] = f2bf(v);
}

__global__ void k_w(const float* __restrict__ x, const float* __restrict__ meta,
                    const float* __restrict__ Wlr, const float* __restrict__ blr,
                    float* __restrict__ w) {
  int wid = threadIdx.x >> 5, lane = threadIdx.x & 31;
  int r = blockIdx.x * 8 + wid;
  if (r >= BT) return;
  int b = r / T, t = r % T;
  const float* src = (t < Mm) ? (meta + (size_t)t * D)
                              : (x + ((size_t)(b * S + (t - Mm))) * D);
  float s = 0.f;
  for (int c = lane; c < D; c += 32) s += src[c] * Wlr[c];
  for (int off = 16; off; off >>= 1) s += __shfl_xor(s, off, 32);
  if (lane == 0) w[r] = MAX_ALR * sigm(s + blr[0]);
}

__global__ void k_dz2(const float* __restrict__ h2, const float* __restrict__ v,
                      const float* __restrict__ z2, const float* __restrict__ w,
                      float* __restrict__ dh2, u16* __restrict__ dz2b) {
  int i = blockIdx.x * 256 + threadIdx.x;
  if (i >= BT * D) return;
  int r = i / D;
  float g = (2.0f / (float)D) * w[r] * (h2[i] - v[i]);
  dh2[i] = g;
  float z = z2[i], s = sigm(z);
  dz2b[i] = f2bf(g * (s * (1.f + z * (1.f - s))));
}

__global__ void k_dz1(const float* __restrict__ dh1, const float* __restrict__ z1,
                      u16* __restrict__ dz1b) {
  int i = blockIdx.x * 256 + threadIdx.x;
  if (i >= BT * D) return;
  float z = z1[i], s = sigm(z);
  dz1b[i] = f2bf(dh1[i] * (s * (1.f + z * (1.f - s))));
}

__global__ void k_colsum(const u16* __restrict__ A, float* __restrict__ out, int rows) {
  int n = blockIdx.x * 256 + threadIdx.x;
  if (n >= D) return;
  float s = 0.f;
  for (int r = 0; r < rows; ++r) s += bf2f(A[(size_t)r * D + n]);
  out[n] = s;
}

// writes updated weight TRANSPOSED: out[n*D+m] = upd(W[m*D+n])  (B^T layout for TN GEMM)
__global__ void k_updW(const float* __restrict__ W, const float* __restrict__ g,
                       u16* __restrict__ outT) {
  int i = blockIdx.x * 256 + threadIdx.x;
  if (i >= D * D) return;
  int m = i / D, n = i % D;
  float gi = g[i];
  float p = W[i] * (1.f - LRc * WDc) - LRc * gi / (fabsf(gi) + EPSc);
  outT[(size_t)n * D + m] = f2bf(p);
}

__global__ void k_updb(const float* __restrict__ bvec, const float* __restrict__ g,
                       float* __restrict__ out) {
  int i = blockIdx.x * 256 + threadIdx.x;
  if (i >= D) return;
  float gi = g[i];
  out[i] = bvec[i] * (1.f - LRc * WDc) - LRc * gi / (fabsf(gi) + EPSc);
}

// ---------------- WMMA TN GEMM with async global->LDS double buffering ----------------
// C[Mrows,N](f32) = epilogue( A @ B + bias ), f32 accumulate.
// A  : [Mrows][K] bf16 row-major (K-major)
// Bt : [N][K]     bf16 row-major (i.e. B transposed; Bt[n][k] = B[k][n])
// mode 0: out = z ;  mode 1: out = resid + silu(z) (z optionally stored to Zout) ;
// mode 2: out = resid + z.   Cb (optional): bf16 copy of out.
// Block: 256 thr = 8 waves, tile 128x128, K-step 32, LDS double buffered,
// staged with GLOBAL_LOAD_ASYNC_TO_LDS_B128 (ASYNCcnt) .
#define LDSTRIDE 40  // u16 units; 80B rows: 16B-aligned b128 frags, conflict-free
__global__ __launch_bounds__(256) void k_gemm(
    const u16* __restrict__ A, const u16* __restrict__ Bt,
    const float* __restrict__ bias, const float* __restrict__ resid,
    float* __restrict__ Cf, u16* __restrict__ Cb, float* __restrict__ Zout,
    int Mrows, int N, int K, int mode) {
  __shared__ u16 As[2 * 128 * LDSTRIDE];
  __shared__ u16 Bs[2 * 128 * LDSTRIDE];

  const int tid = threadIdx.x;
  const int w = tid >> 5, lane = tid & 31;
  const int lm = lane & 15, hh = lane >> 4;
  const int wm = w >> 1, wn = w & 1;                 // 4x2 wave grid
  const int m0 = blockIdx.y * 128, n0 = blockIdx.x * 128;

  v8f acc[8];
#pragma unroll
  for (int i = 0; i < 8; ++i)
#pragma unroll
    for (int r = 0; r < 8; ++r) acc[i][r] = 0.f;

  const unsigned lABase = (unsigned)(size_t)(&As[0]);
  const unsigned lBBase = (unsigned)(size_t)(&Bs[0]);

  auto stage = [&](int buf, int k0) {
    unsigned lA = lABase + (unsigned)(buf * 128 * LDSTRIDE * 2);
    unsigned lB = lBBase + (unsigned)(buf * 128 * LDSTRIDE * 2);
#pragma unroll
    for (int j = 0; j < 2; ++j) {
      int c = tid + j * 256;            // 512 16B-chunks per tile
      int row = c >> 2, ch = c & 3;
      unsigned loff = (unsigned)(row * (LDSTRIDE * 2) + ch * 16);
      const u16* gA = A + (size_t)(m0 + row) * K + (k0 + ch * 8);
      unsigned dA = lA + loff;
      asm volatile("global_load_async_to_lds_b128 %0, %1, off"
                   :: "v"(dA), "v"(gA) : "memory");
      const u16* gB = Bt + (size_t)(n0 + row) * K + (k0 + ch * 8);
      unsigned dB = lB + loff;
      asm volatile("global_load_async_to_lds_b128 %0, %1, off"
                   :: "v"(dB), "v"(gB) : "memory");
    }
  };

  stage(0, 0);
  const int nk = K >> 5;
  for (int i = 0; i < nk; ++i) {
    asm volatile("s_wait_asynccnt 0x0" ::: "memory");
    __syncthreads();                       // publishes buf (i&1); proves buf ((i+1)&1) free
    if (i + 1 < nk) stage((i + 1) & 1, (i + 1) * 32);

    const u16* sA = As + (i & 1) * 128 * LDSTRIDE;
    const u16* sB = Bs + (i & 1) * 128 * LDSTRIDE;
    union F { v16bf v; uint4 q[2]; };
    F a[2];
#pragma unroll
    for (int mt = 0; mt < 2; ++mt) {
      // A frag: lane row m=lm; elems 0..7 -> k=8*hh.., elems 8..15 -> k=16+8*hh..
      const u16* ap = sA + (wm * 32 + mt * 16 + lm) * LDSTRIDE;
      a[mt].q[0] = *(const uint4*)(ap + 8 * hh);
      a[mt].q[1] = *(const uint4*)(ap + 16 + 8 * hh);
    }
#pragma unroll
    for (int ct = 0; ct < 4; ++ct) {
      // B frag: lane col n=lm; elems 0..7 -> k=8*hh.., elems 8..15 -> k=16+8*hh..
      F bfr;
      const u16* bp = sB + (wn * 64 + ct * 16 + lm) * LDSTRIDE;
      bfr.q[0] = *(const uint4*)(bp + 8 * hh);
      bfr.q[1] = *(const uint4*)(bp + 16 + 8 * hh);
#pragma unroll
      for (int mt = 0; mt < 2; ++mt)
        acc[mt * 4 + ct] = __builtin_amdgcn_wmma_f32_16x16x32_bf16(
            false, a[mt].v, false, bfr.v, (short)0, acc[mt * 4 + ct], false, false);
    }
  }

#pragma unroll
  for (int mt = 0; mt < 2; ++mt)
#pragma unroll
    for (int ct = 0; ct < 4; ++ct) {
      int col = n0 + wn * 64 + ct * 16 + lm;
      float bv = bias ? bias[col] : 0.f;
#pragma unroll
      for (int r = 0; r < 8; ++r) {
        int row = m0 + wm * 32 + mt * 16 + r + 8 * hh;  // C layout: m = r + 8*hh
        size_t idx = (size_t)row * N + col;
        float z = acc[mt * 4 + ct][r] + bv;
        float outv;
        if (mode == 1) {
          if (Zout) Zout[idx] = z;
          outv = resid[idx] + z * sigm(z);
        } else if (mode == 2) {
          outv = resid[idx] + z;
        } else {
          outv = z;
        }
        Cf[idx] = outv;
        if (Cb) Cb[idx] = f2bf(outv);
      }
    }
}

// ---------------- sliding-window attention (online softmax) ----------------
__global__ __launch_bounds__(256) void k_attn(const float* __restrict__ rq,
                                              const float* __restrict__ rk,
                                              const float* __restrict__ rv,
                                              u16* __restrict__ outb) {
  int bx = blockIdx.x, tid = threadIdx.x;
  int b = bx >> 5, h = (bx >> 2) & 7, sc = bx & 3;  // 4*8*4 = 128 blocks
  int s = sc * 256 + tid, t = Mm + s;
  const float* qp = rq + ((size_t)(b * T + t)) * D + h * HD;
  float qv[HD], acc[HD];
#pragma unroll
  for (int d = 0; d < HD; ++d) { qv[d] = qp[d] * 0.125f; acc[d] = 0.f; }  // 1/sqrt(64)
  float mmax = -3.0e38f, denom = 0.f;
  int j0 = t - WIN + 1; if (j0 < 0) j0 = 0;
  for (int j = j0; j <= t; ++j) {
    const float* kp = rk + ((size_t)(b * T + j)) * D + h * HD;
    float dot = 0.f;
#pragma unroll
    for (int d = 0; d < HD; ++d) dot += qv[d] * kp[d];
    float nm = fmaxf(mmax, dot);
    float corr = __expf(mmax - nm);
    float p = __expf(dot - nm);
    denom = denom * corr + p;
    const float* vp = rv + ((size_t)(b * T + j)) * D + h * HD;
#pragma unroll
    for (int d = 0; d < HD; ++d) acc[d] = acc[d] * corr + p * vp[d];
    mmax = nm;
  }
  float inv = 1.f / denom;
  u16* op = outb + ((size_t)(b * S + s)) * D + h * HD;
#pragma unroll
  for (int d = 0; d < HD; ++d) op[d] = f2bf(acc[d] * inv);
}

// ---------------- host launch ----------------

extern "C" void kernel_launch(void* const* d_in, const int* in_sizes, int n_in,
                              void* d_out, int out_size, void* d_ws, size_t ws_size,
                              hipStream_t stream) {
  (void)in_sizes; (void)n_in; (void)out_size; (void)ws_size;
  const float* x    = (const float*)d_in[0];
  const float* meta = (const float*)d_in[1];
  const float* lmmW = (const float*)d_in[2];
  const float* lmmb = (const float*)d_in[3];
  const float* Wq = (const float*)d_in[4];  const float* bq = (const float*)d_in[5];
  const float* Wk = (const float*)d_in[6];  const float* bk = (const float*)d_in[7];
  const float* Wv = (const float*)d_in[8];  const float* bv = (const float*)d_in[9];
  const float* Wlr = (const float*)d_in[10]; const float* blr = (const float*)d_in[11];
  const float* swaW = (const float*)d_in[12]; const float* swab = (const float*)d_in[13];

  char* ws = (char*)d_ws;
  size_t off = 0;
  auto alloc = [&](size_t bytes) -> char* {
    char* p = ws + off;
    off = (off + bytes + 255) & ~(size_t)255;
    return p;
  };
  const size_t BTD = (size_t)BT * D, DD = (size_t)D * D;

  // bf16 data buffers
  u16* xmb   = (u16*)alloc(BTD * 2);
  u16* qb    = (u16*)alloc(BTD * 2);
  u16* kb    = (u16*)alloc(BTD * 2);
  u16* kbT   = (u16*)alloc(BTD * 2);
  u16* h1kb  = (u16*)alloc(BTD * 2);
  u16* h1kbT = (u16*)alloc(BTD * 2);
  u16* dz2b  = (u16*)alloc(BTD * 2);
  u16* dz2bT = (u16*)alloc(BTD * 2);
  u16* dz1b  = (u16*)alloc(BTD * 2);
  u16* dz1bT = (u16*)alloc(BTD * 2);
  // bf16 weights, all stored as B^T ([out][in]) unless noted
  u16* wqT  = (u16*)alloc(DD * 2);
  u16* wkT  = (u16*)alloc(DD * 2);
  u16* wvT  = (u16*)alloc(DD * 2);
  u16* w0T  = (u16*)alloc(DD * 2);
  u16* w1T  = (u16*)alloc(DD * 2);
  u16* w1p  = (u16*)alloc(DD * 2);   // W1 plain ([in][out]) == (W1^T)^T, Bt for dh1 GEMM
  u16* wn0T = (u16*)alloc(DD * 2);
  u16* wn1T = (u16*)alloc(DD * 2);
  u16* swaT = (u16*)alloc(4 * DD * 2);
  // f32 buffers
  float* qf   = (float*)alloc(BTD * 4);
  float* kf   = (float*)alloc(BTD * 4);
  float* vf   = (float*)alloc(BTD * 4);
  float* z1kf = (float*)alloc(BTD * 4);
  float* h1kf = (float*)alloc(BTD * 4);
  float* z2kf = (float*)alloc(BTD * 4);
  float* h2kf = (float*)alloc(BTD * 4);
  float* dh2f = (float*)alloc(BTD * 4);
  float* dh1f = (float*)alloc(BTD * 4);
  float* wts  = (float*)alloc((size_t)BT * 4);
  float* gW0f = (float*)alloc(DD * 4);
  float* gW1f = (float*)alloc(DD * 4);
  float* gb0f = (float*)alloc((size_t)D * 4);
  float* gb1f = (float*)alloc((size_t)D * 4);
  float* b0nf = (float*)alloc((size_t)D * 4);
  float* b1nf = (float*)alloc((size_t)D * 4);

  // lifetime-safe aliases
  u16* h1qb   = dz2b;   // dz2b dead after dh1/colsum/transpose (pre-update)
  u16* retrb  = dz1b;   // dz1b dead after colsum/transpose (pre-update)
  u16* attncb = xmb;    // xmb dead after q/k/v (B*S*D <= BT*D)
  float* h1qf = h1kf;   // dead after z2k gemm
  float* retrf = z1kf;  // dead after dz1
  float* rqf  = h2kf;   // dead after dz2
  float* rkf  = dh2f;   // dead after dh1 gemm
  float* rvf  = dh1f;   // dead after dz1

  dim3 b256(256);
  auto g1 = [&](size_t n) { return dim3((unsigned)((n + 255) / 256)); };
  auto GG = [&](int Mr) { return dim3(D / 128, (unsigned)(Mr / 128)); };
  dim3 trg(D / 32, BT / 32);

  // weight conversions (transposed bf16 = B^T layout for TN GEMM)
  k_cvtT<<<g1(DD), b256, 0, stream>>>(Wq, wqT);
  k_cvtT<<<g1(DD), b256, 0, stream>>>(Wk, wkT);
  k_cvtT<<<g1(DD), b256, 0, stream>>>(Wv, wvT);
  k_cvtT<<<g1(DD), b256, 0, stream>>>(lmmW, w0T);
  k_cvtT<<<g1(DD), b256, 0, stream>>>(lmmW + DD, w1T);
  k_cvt<<<g1(DD), b256, 0, stream>>>(lmmW + DD, w1p, (int)DD);
  for (int i = 0; i < 4; ++i)
    k_cvtT<<<g1(DD), b256, 0, stream>>>(swaW + i * DD, swaT + i * DD);
  // meta-token injection + adaptive lr
  k_build_xm<<<g1(BTD), b256, 0, stream>>>(x, meta, xmb);
  k_w<<<dim3(BT / 8), b256, 0, stream>>>(x, meta, Wlr, blr, wts);

  // q, k, v projections
  k_gemm<<<GG(BT), b256, 0, stream>>>(xmb, wqT, bq, nullptr, qf, qb, nullptr, BT, D, D, 0);
  k_gemm<<<GG(BT), b256, 0, stream>>>(xmb, wkT, bk, nullptr, kf, kb, nullptr, BT, D, D, 0);
  k_gemm<<<GG(BT), b256, 0, stream>>>(xmb, wvT, bv, nullptr, vf, nullptr, nullptr, BT, D, D, 0);
  k_tr<<<trg, b256, 0, stream>>>(kb, kbT, BT, D);

  // ResLinear forward on k
  k_gemm<<<GG(BT), b256, 0, stream>>>(kb, w0T, lmmb, kf, h1kf, h1kb, z1kf, BT, D, D, 1);
  k_tr<<<trg, b256, 0, stream>>>(h1kb, h1kbT, BT, D);
  k_gemm<<<GG(BT), b256, 0, stream>>>(h1kb, w1T, lmmb + D, h1kf, h2kf, nullptr, z2kf, BT, D, D, 1);

  // backward
  k_dz2<<<g1(BTD), b256, 0, stream>>>(h2kf, vf, z2kf, wts, dh2f, dz2b);
  k_tr<<<trg, b256, 0, stream>>>(dz2b, dz2bT, BT, D);
  k_gemm<<<GG(BT), b256, 0, stream>>>(dz2b, w1p, nullptr, dh2f, dh1f, nullptr, nullptr, BT, D, D, 2);
  k_colsum<<<g1(D), b256, 0, stream>>>(dz2b, gb1f, BT);
  k_gemm<<<GG(512), b256, 0, stream>>>(h1kbT, dz2bT, nullptr, nullptr, gW1f, nullptr, nullptr, 512, D, BT, 0);
  k_dz1<<<g1(BTD), b256, 0, stream>>>(dh1f, z1kf, dz1b);
  k_tr<<<trg, b256, 0, stream>>>(dz1b, dz1bT, BT, D);
  k_colsum<<<g1(D), b256, 0, stream>>>(dz1b, gb0f, BT);
  k_gemm<<<GG(512), b256, 0, stream>>>(kbT, dz1bT, nullptr, nullptr, gW0f, nullptr, nullptr, 512, D, BT, 0);

  // AdamW-first-step style update (weights written transposed = B^T layout)
  k_updW<<<g1(DD), b256, 0, stream>>>(lmmW, gW0f, wn0T);
  k_updW<<<g1(DD), b256, 0, stream>>>(lmmW + DD, gW1f, wn1T);
  k_updb<<<g1(D), b256, 0, stream>>>(lmmb, gb0f, b0nf);
  k_updb<<<g1(D), b256, 0, stream>>>(lmmb + D, gb1f, b1nf);

  // retrieval: ResLinear with updated params on q
  k_gemm<<<GG(BT), b256, 0, stream>>>(qb, wn0T, b0nf, qf, h1qf, h1qb, nullptr, BT, D, D, 1);
  k_gemm<<<GG(BT), b256, 0, stream>>>(h1qb, wn1T, b1nf, h1qf, retrf, retrb, nullptr, BT, D, D, 1);

  // attention projections
  k_gemm<<<GG(BT), b256, 0, stream>>>(retrb, swaT, swab, nullptr, rqf, nullptr, nullptr, BT, D, D, 0);
  k_gemm<<<GG(BT), b256, 0, stream>>>(retrb, swaT + DD, swab + D, nullptr, rkf, nullptr, nullptr, BT, D, D, 0);
  k_gemm<<<GG(BT), b256, 0, stream>>>(retrb, swaT + 2 * DD, swab + 2 * D, nullptr, rvf, nullptr, nullptr, BT, D, D, 0);

  // sliding-window attention over non-meta rows -> bf16 compact buffer
  k_attn<<<dim3(128), b256, 0, stream>>>(rqf, rkf, rvf, attncb);

  // final output projection straight into d_out [B*S, D] f32
  k_gemm<<<GG(Bsz * S), b256, 0, stream>>>(attncb, swaT + 3 * DD, swab + 3 * D, nullptr,
                                           (float*)d_out, nullptr, nullptr, Bsz * S, D, D, 0);
}
